// LinearRNN_41506563948812
// MI455X (gfx1250) — matrix operand order repaired and checked by reading, not compiled, not run on hardware
//
#include <hip/hip_runtime.h>
#include <hip/hip_bf16.h>

// ---------------------------------------------------------------------------
// Linear RNN on MI455X (gfx1250): chunked parallel scan, WMMA f16 (f32 acc),
// async global->LDS staging of the x tile, pre-transposed f16 weights so every
// B fragment is two contiguous b128 loads.
//   h_t = x_t@W_ih + b + h_{t-1}@W_hh ;  y_t = h_t@W_ho + b_ho
// Serial depth 8192 -> 64 (local scans) + 128 (carry scan) + 64 (replay).
// ---------------------------------------------------------------------------

typedef _Float16 half16 __attribute__((ext_vector_type(16)));
typedef _Float16 half8v __attribute__((ext_vector_type(8)));
typedef float    float8 __attribute__((ext_vector_type(8)));
typedef int      i4v    __attribute__((ext_vector_type(4)));

#define T_LEN  8192
#define BATCH  16
#define IDIM   64
#define HDIM   256
#define ODIM   64
#define CHUNK  64
#define NCHUNK 128            // T_LEN / CHUNK
#define LDS_LD 264            // padded f16 row stride for h (bank-conflict free)
#define XROW_LD 88            // padded f16 row stride for staged x (16B-mult, c.f.)
#define SMEM_BYTES ((BATCH * LDS_LD + CHUNK * BATCH * XROW_LD) * 2)  // ~188KB

// gfx1250 async global->LDS path (guarded: falls back to sync copy if absent)
#if defined(__HIP_DEVICE_COMPILE__) &&                                        \
    __has_builtin(__builtin_amdgcn_global_load_async_to_lds_b128) &&          \
    __has_builtin(__builtin_amdgcn_s_wait_asynccnt)
#define ASYNC_OK 1
// Builtin prototype (from hipcc diagnostic): (int4 AS1*, int4 AS3*, imm, imm)
typedef __attribute__((address_space(1))) i4v* g_i4p;
typedef __attribute__((address_space(3))) i4v* l_i4p;
#else
#define ASYNC_OK 0
#endif

// --- WMMA wrapper (codegen-confirmed builtin) ------------------------------
__device__ __forceinline__ float8 wmma_f16(half16 a, half16 b, float8 c) {
  return __builtin_amdgcn_wmma_f32_16x16x32_f16(false, a, false, b,
                                                (short)0, c, false, false);
}

// --- A fragment: 16x32 f16 from a row-major row pointer.
// lane m holds row m; element i -> k = kc*32 + 16*kh + i (matches B)
// => one contiguous 32B run per lane.
__device__ __forceinline__ half16 ld_a_frag(const _Float16* rowp, int kc, int kh) {
  half8v lo = *(const half8v*)(rowp + kc * 32 + 16 * kh);
  half8v hi = *(const half8v*)(rowp + kc * 32 + 16 * kh + 8);
  return __builtin_shufflevector(lo, hi, 0, 1, 2, 3, 4, 5, 6, 7,
                                 8, 9, 10, 11, 12, 13, 14, 15);
}

// --- B fragment: 32x16 f16 from TRANSPOSED weight Wt[n][k] (ld = ldk).
// lane: n = lane%16, kh = lane/16; element i -> k = kbase + 16*kh + i
// => one contiguous 32B run per lane (two b128 loads).
__device__ __forceinline__ half16 ld_b_frag_t(const _Float16* Wt, int ldk,
                                              int kbase, int ncol, int kh) {
  const _Float16* p = Wt + (size_t)ncol * ldk + kbase + 16 * kh;
  half8v lo = *(const half8v*)(p);
  half8v hi = *(const half8v*)(p + 8);
  return __builtin_shufflevector(lo, hi, 0, 1, 2, 3, 4, 5, 6, 7,
                                 8, 9, 10, 11, 12, 13, 14, 15);
}

// --- f32 -> f16 elementwise --------------------------------------------------
__global__ void cvt_f32_to_f16_kernel(const float* __restrict__ in,
                                      _Float16* __restrict__ out, int n) {
  int i = blockIdx.x * 256 + threadIdx.x;
  if (i < n) out[i] = (_Float16)in[i];
}

// --- f32 [K][N] -> f16 transposed [N][K] ------------------------------------
__global__ void cvt_transpose_kernel(const float* __restrict__ in,
                                     _Float16* __restrict__ out, int K, int N) {
  int i = blockIdx.x * 256 + threadIdx.x;
  if (i < N * K) {
    int n = i / K, k = i - n * K;
    out[i] = (_Float16)in[k * N + n];
  }
}

// --- out = in @ in (256x256, f32 acc); emits row-major AND transposed -------
__global__ __launch_bounds__(512) void matsq_kernel(
    const _Float16* __restrict__ inA,   // row-major [256][256]
    const _Float16* __restrict__ inT,   // transposed [256][256]
    _Float16* __restrict__ outA, _Float16* __restrict__ outT) {
  const int tid = threadIdx.x;
  const int w = tid >> 5, l = tid & 31;
  const int kh = l >> 4, lm = l & 15;
  const int ncol = (w << 4) + lm;
  const int bm = blockIdx.x;  // 0..15 row tile

  const _Float16* arow = inA + (size_t)(bm * 16 + lm) * HDIM;
  float8 acc;
#pragma unroll
  for (int j = 0; j < 8; ++j) acc[j] = 0.f;
#pragma unroll
  for (int kc = 0; kc < 8; ++kc)
    acc = wmma_f16(ld_a_frag(arow, kc, kh),
                   ld_b_frag_t(inT, HDIM, kc * 32, ncol, kh), acc);
#pragma unroll
  for (int j = 0; j < 8; ++j) {
    const int mrow = bm * 16 + j + 8 * kh;
    _Float16 v = (_Float16)acc[j];
    outA[(size_t)mrow * HDIM + ncol] = v;
    outT[(size_t)ncol * HDIM + mrow] = v;
  }
}

// --- chunk scan (pass1: emit chunk-end e_c; pass2: replay w/ carry, emit h) -
template <bool WRITE_H>
__global__ __launch_bounds__(512) void rnn_pass_kernel(
    const _Float16* __restrict__ x16,      // [B,T,I] f16
    const _Float16* __restrict__ WihT,     // [H][I]  f16 (transposed)
    const _Float16* __restrict__ WhhT,     // [H][H]  f16 (transposed)
    const float* __restrict__ b_ih, const float* __restrict__ b_hh,
    const float* __restrict__ carry_in,    // [NC,B,H] f32 (pass2: a[c-1])
    float* __restrict__ e_out,             // [NC,B,H] f32 (pass1)
    _Float16* __restrict__ h_all) {        // [T,B,H]  f16 (pass2)
  extern __shared__ _Float16 smem[];
  _Float16* hbuf = smem;                       // [16][LDS_LD]
  _Float16* xbuf = smem + BATCH * LDS_LD;      // [CHUNK*16][XROW_LD]

  const int c = blockIdx.x;                // chunk id
  const int tid = threadIdx.x;
  const int w = tid >> 5, l = tid & 31;
  const int kh = l >> 4, lm = l & 15;
  const int ncol = (w << 4) + lm;          // wave owns output cols [16w,16w+16)

  // ---- Stage this chunk's whole x tile into LDS (async DMA, overlaps the
  // resident-B prologue loads below). 8192 b128 segments / 512 threads.
#pragma unroll
  for (int it = 0; it < 16; ++it) {
    const int j = tid + it * 512;          // 0..8191
    const int seg = j & 7;                 // 8 x 16B per 128B x-row
    const int row = j >> 3;                // 0..1023 ; row = tl*16 + m
    const int m = row & 15;
    const int tl = row >> 4;
    const _Float16* src = x16 + (size_t)m * (T_LEN * IDIM) +
                          (size_t)(c * CHUNK + tl) * IDIM + seg * 8;
    _Float16* dst = xbuf + (size_t)row * XROW_LD + seg * 8;
#if ASYNC_OK
    __builtin_amdgcn_global_load_async_to_lds_b128((g_i4p)src, (l_i4p)dst, 0, 0);
#else
    *(half8v*)dst = *(const half8v*)src;
#endif
  }

  // ---- Resident B fragments (contiguous b128 loads from transposed weights)
  half16 bw[8];
#pragma unroll
  for (int kc = 0; kc < 8; ++kc)
    bw[kc] = ld_b_frag_t(WhhT, HDIM, kc * 32, ncol, kh);
  half16 bx[2];
#pragma unroll
  for (int kc = 0; kc < 2; ++kc)
    bx[kc] = ld_b_frag_t(WihT, IDIM, kc * 32, ncol, kh);
  const float bias = b_ih[ncol] + b_hh[ncol];

  // ---- h0: zero (pass1 / chunk 0) or carry a[c-1] (pass2, c>0)
  if (WRITE_H && c > 0) {
    const float* src = carry_in + (size_t)(c - 1) * (BATCH * HDIM);
    for (int i = tid; i < BATCH * HDIM; i += 512)
      hbuf[(i >> 8) * LDS_LD + (i & 255)] = (_Float16)src[i];
  } else {
    for (int i = tid; i < BATCH * HDIM; i += 512)
      hbuf[(i >> 8) * LDS_LD + (i & 255)] = (_Float16)0.f;
  }
#if ASYNC_OK
  __builtin_amdgcn_s_wait_asynccnt(0);     // own DMA done; barrier makes global
#endif
  __syncthreads();

  // ---- serial scan: every step is pure LDS + WMMA
  const _Float16* hrow = &hbuf[lm * LDS_LD];   // A row = batch = lane%16
  float8 acc;
  for (int s = 0; s < CHUNK; ++s) {
#pragma unroll
    for (int j = 0; j < 8; ++j) acc[j] = bias;

    const _Float16* xrow = xbuf + (size_t)(s * 16 + lm) * XROW_LD;
#pragma unroll
    for (int kc = 0; kc < 2; ++kc)          // u_t = x_t @ W_ih (K=64)
      acc = wmma_f16(ld_a_frag(xrow, kc, kh), bx[kc], acc);
#pragma unroll
    for (int kc = 0; kc < 8; ++kc)          // + h_{t-1} @ W_hh (K=256)
      acc = wmma_f16(ld_a_frag(hrow, kc, kh), bw[kc], acc);

    __syncthreads();  // WAR: all waves done reading h_{t-1}
#pragma unroll
    for (int j = 0; j < 8; ++j)             // C layout: m = j + 8*kh
      hbuf[(j + 8 * kh) * LDS_LD + ncol] = (_Float16)acc[j];
    if (WRITE_H) {
      _Float16* hg = h_all + (size_t)(c * CHUNK + s) * (BATCH * HDIM);
#pragma unroll
      for (int j = 0; j < 8; ++j)
        hg[(j + 8 * kh) * HDIM + ncol] = (_Float16)acc[j];
    }
    __syncthreads();  // RAW: h_t visible for next step
  }

  if (!WRITE_H) {     // pass1: chunk-end value straight from f32 accumulators
    float* eg = e_out + (size_t)c * (BATCH * HDIM);
#pragma unroll
    for (int j = 0; j < 8; ++j)
      eg[(j + 8 * kh) * HDIM + ncol] = acc[j];
  }
}

// --- chunk-level carry scan: a_c = e_c + a_{c-1} @ W^64 (128 steps, 1 WG) ---
__global__ __launch_bounds__(512) void carry_scan_kernel(
    const _Float16* __restrict__ WcT,  // (W_hh^64)^T f16
    const float* __restrict__ e,       // [NC,B,H]
    float* __restrict__ aout) {        // [NC,B,H]
  __shared__ _Float16 hbuf[BATCH * LDS_LD];
  const int tid = threadIdx.x;
  const int w = tid >> 5, l = tid & 31;
  const int kh = l >> 4, lm = l & 15;
  const int ncol = (w << 4) + lm;

  half16 bw[8];
#pragma unroll
  for (int kc = 0; kc < 8; ++kc)
    bw[kc] = ld_b_frag_t(WcT, HDIM, kc * 32, ncol, kh);

  for (int i = tid; i < BATCH * HDIM; i += 512)
    hbuf[(i >> 8) * LDS_LD + (i & 255)] = (_Float16)0.f;
  __syncthreads();

  const _Float16* hrow = &hbuf[lm * LDS_LD];
  for (int c = 0; c < NCHUNK; ++c) {
    const float* eg = e + (size_t)c * (BATCH * HDIM);
    float8 acc;
#pragma unroll
    for (int j = 0; j < 8; ++j) acc[j] = eg[(j + 8 * kh) * HDIM + ncol];
#pragma unroll
    for (int kc = 0; kc < 8; ++kc)
      acc = wmma_f16(ld_a_frag(hrow, kc, kh), bw[kc], acc);
    __syncthreads();
    float* ag = aout + (size_t)c * (BATCH * HDIM);
#pragma unroll
    for (int j = 0; j < 8; ++j) {
      hbuf[(j + 8 * kh) * LDS_LD + ncol] = (_Float16)acc[j];
      ag[(j + 8 * kh) * HDIM + ncol] = acc[j];
    }
    __syncthreads();
  }
}

// --- y = h @ W_ho + b_ho ([T*16,256]@[256,64], fully parallel) --------------
__global__ __launch_bounds__(512) void yproj_kernel(
    const _Float16* __restrict__ h_all,  // [T,B,H] f16
    const _Float16* __restrict__ WhoT,   // [O][H]  f16 (transposed)
    const float* __restrict__ b_ho,      // [O]
    float* __restrict__ y) {             // [B,T,O] f32
  const int tid = threadIdx.x;
  const int w = tid >> 5, l = tid & 31;
  const int kh = l >> 4, lm = l & 15;
  const int nt = w & 3;                  // O = 4 x 16 col tiles
  const int tg = w >> 2;
  const int ncol = nt * 16 + lm;

  half16 bw[8];
#pragma unroll
  for (int kc = 0; kc < 8; ++kc)
    bw[kc] = ld_b_frag_t(WhoT, HDIM, kc * 32, ncol, kh);
  const float bias = b_ho[ncol];

  for (int tt = 0; tt < 16; ++tt) {
    const int t = blockIdx.x * 64 + tg * 16 + tt;
    const _Float16* arow =
        h_all + (size_t)t * (BATCH * HDIM) + (size_t)lm * HDIM;
    __builtin_prefetch(arow + BATCH * HDIM);   // global_prefetch_b8: next t
    float8 acc;
#pragma unroll
    for (int j = 0; j < 8; ++j) acc[j] = bias;
#pragma unroll
    for (int kc = 0; kc < 8; ++kc)
      acc = wmma_f16(ld_a_frag(arow, kc, kh), bw[kc], acc);
#pragma unroll
    for (int j = 0; j < 8; ++j)          // y[b][t][o]
      y[(size_t)(j + 8 * kh) * (T_LEN * ODIM) + (size_t)t * ODIM + ncol] =
          acc[j];
  }
}

// ---------------------------------------------------------------------------
extern "C" void kernel_launch(void* const* d_in, const int* in_sizes, int n_in,
                              void* d_out, int out_size, void* d_ws,
                              size_t ws_size, hipStream_t stream) {
  (void)in_sizes; (void)n_in; (void)out_size; (void)ws_size;
  const float* x    = (const float*)d_in[0];
  const float* W_ih = (const float*)d_in[1];
  const float* W_hh = (const float*)d_in[2];
  const float* b_ih = (const float*)d_in[3];
  const float* b_hh = (const float*)d_in[4];
  const float* W_ho = (const float*)d_in[5];
  const float* b_ho = (const float*)d_in[6];
  float* y = (float*)d_out;

  // Workspace carve (~88 MB total)
  char* ws = (char*)d_ws;
  size_t off = 0;
  auto carve = [&](size_t bytes) -> void* {
    void* p = ws + off;
    off += (bytes + 255) & ~(size_t)255;
    return p;
  };
  _Float16* x16   = (_Float16*)carve((size_t)BATCH * T_LEN * IDIM * 2);  // 16.8MB
  _Float16* h_all = (_Float16*)carve((size_t)T_LEN * BATCH * HDIM * 2);  // 67MB
  _Float16* WihT  = (_Float16*)carve((size_t)IDIM * HDIM * 2);
  _Float16* Whh16 = (_Float16*)carve((size_t)HDIM * HDIM * 2);
  _Float16* WhhT  = (_Float16*)carve((size_t)HDIM * HDIM * 2);
  _Float16* WhoT  = (_Float16*)carve((size_t)HDIM * ODIM * 2);
  _Float16* pA    = (_Float16*)carve((size_t)HDIM * HDIM * 2);
  _Float16* pAT   = (_Float16*)carve((size_t)HDIM * HDIM * 2);
  _Float16* pB    = (_Float16*)carve((size_t)HDIM * HDIM * 2);
  _Float16* pBT   = (_Float16*)carve((size_t)HDIM * HDIM * 2);
  float*    e_buf = (float*)carve((size_t)NCHUNK * BATCH * HDIM * 4);    // 2MB
  float*    a_buf = (float*)carve((size_t)NCHUNK * BATCH * HDIM * 4);    // 2MB

  // 1) conversions: x (row-major f16), weights (transposed f16; W_hh also RM)
  int nx = BATCH * T_LEN * IDIM;
  cvt_f32_to_f16_kernel<<<(nx + 255) / 256, 256, 0, stream>>>(x, x16, nx);
  cvt_f32_to_f16_kernel<<<(HDIM * HDIM + 255) / 256, 256, 0, stream>>>(
      W_hh, Whh16, HDIM * HDIM);
  cvt_transpose_kernel<<<(HDIM * HDIM + 255) / 256, 256, 0, stream>>>(
      W_hh, WhhT, HDIM, HDIM);
  cvt_transpose_kernel<<<(IDIM * HDIM + 255) / 256, 256, 0, stream>>>(
      W_ih, WihT, IDIM, HDIM);
  cvt_transpose_kernel<<<(HDIM * ODIM + 255) / 256, 256, 0, stream>>>(
      W_ho, WhoT, HDIM, ODIM);

  // 2) W_hh^64 via 6 squarings (each emits row-major + transposed)
  matsq_kernel<<<16, 512, 0, stream>>>(Whh16, WhhT, pA, pAT);  // W^2
  matsq_kernel<<<16, 512, 0, stream>>>(pA, pAT, pB, pBT);      // W^4
  matsq_kernel<<<16, 512, 0, stream>>>(pB, pBT, pA, pAT);      // W^8
  matsq_kernel<<<16, 512, 0, stream>>>(pA, pAT, pB, pBT);      // W^16
  matsq_kernel<<<16, 512, 0, stream>>>(pB, pBT, pA, pAT);      // W^32
  matsq_kernel<<<16, 512, 0, stream>>>(pA, pAT, pB, pBT);      // W^64

  // 3) pass 1: local scans with h0 = 0, emit chunk-end e_c
  rnn_pass_kernel<false><<<NCHUNK, 512, SMEM_BYTES, stream>>>(
      x16, WihT, WhhT, b_ih, b_hh, nullptr, e_buf, nullptr);

  // 4) carry scan: a_c = e_c + a_{c-1} @ W^64
  carry_scan_kernel<<<1, 512, 0, stream>>>(pBT, e_buf, a_buf);

  // 5) pass 2: replay with correct carries, emit all h_t
  rnn_pass_kernel<true><<<NCHUNK, 512, SMEM_BYTES, stream>>>(
      x16, WihT, WhhT, b_ih, b_hh, a_buf, nullptr, h_all);

  // 6) output projection
  yproj_kernel<<<NCHUNK, 512, 0, stream>>>(h_all, WhoT, b_ho, y);
}
